// GeneratorGeoECGAN_41583873360606
// MI455X (gfx1250) — compile-verified, matrix operand order checked by reading, not run on hardware
//
#include <hip/hip_runtime.h>

// ---------------------------------------------------------------------------
// GeneratorGeoECGAN for MI455X (gfx1250): bf16 WMMA GEMM pipeline
// B=16384, R=360, NM=200, H=256, P1=512, G1=256, RC1=256, LAT=128
// ---------------------------------------------------------------------------

#define Bsz  16384
#define Rr   360
#define NMm  200
#define Hh   256
#define P1p  512
#define LATl 128
#define K3R  1080          // 3*R
#define RRf  (Rr*Rr)       // 129600

typedef __attribute__((ext_vector_type(16))) __bf16       bf16x16;
typedef __attribute__((ext_vector_type(2)))  __bf16       bf16x2;
typedef __attribute__((ext_vector_type(8)))  float        floatx8;
typedef __attribute__((ext_vector_type(8)))  unsigned int uintx8;

#define BM 64
#define BN 64
#define BK 32
// 36 shorts = 72B row stride: 8B-aligned rows (ds_store_b64 legal) and an
// 18-bank stride -> 16 distinct banks across a 16-row fragment read group.
#define LDS_STRIDE 36

#define FLAG_RELU   1
#define FLAG_STOREC 2
#define FLAG_MASK   4

// Hardware packed f32->bf16 convert (v_cvt_pk_bf16_f32).
__device__ __forceinline__ unsigned int pack_bf16x2(float a, float b) {
  bf16x2 v = { (__bf16)a, (__bf16)b };
  return __builtin_bit_cast(unsigned int, v);
}
__device__ __forceinline__ unsigned short f32_to_bf16(float f) {
  return __builtin_bit_cast(unsigned short, (__bf16)f);
}

// Gather one 16x32 bf16 WMMA operand fragment from an LDS tile laid out
// [rows][LDS_STRIDE] (K fast). Layout per ISA 7.12.2: lane<16 holds
// K in {0..7,16..23}, lane>=16 (hi=1) holds K in {8..15,24..31};
// VGPR p holds an even/odd K pair -> read as aligned dwords.
__device__ __forceinline__ bf16x16 load_frag(const unsigned short* tile, int row, int hi) {
  const unsigned int* p32 = (const unsigned int*)(tile + row * LDS_STRIDE);
  uintx8 raw;
#pragma unroll
  for (int p = 0; p < 8; ++p) {
    int k0 = (p < 4) ? (2 * p + hi * 8) : (16 + 2 * (p - 4) + hi * 8);
    raw[p] = p32[k0 >> 1];
  }
  return __builtin_bit_cast(bf16x16, raw);
}

// ---- software-pipelined staging: load phase (global -> regs) ---------------
__device__ __forceinline__ void load_phase(
    const float* __restrict__ A, const float* __restrict__ W,
    int K, int rowBase, int colBase, int kBase, int tid,
    float4* ra, float4* rb) {
#pragma unroll
  for (int i = 0; i < 4; ++i) {
    int q = tid + i * 128;              // float4 index 0..511
    int r = q >> 3, c4 = (q & 7) << 2;
    ra[i] = *(const float4*)(A + (long)(rowBase + r) * K + kBase + c4);
    rb[i] = *(const float4*)(W + (long)(colBase + r) * K + kBase + c4);
  }
}

// ---- store phase (regs -> packed bf16 in LDS) ------------------------------
__device__ __forceinline__ void store_phase(
    int tid, const float4* ra, const float4* rb,
    unsigned short (*As)[LDS_STRIDE], unsigned short (*Bs)[LDS_STRIDE]) {
#pragma unroll
  for (int i = 0; i < 4; ++i) {
    int q = tid + i * 128;
    int r = q >> 3, c4 = (q & 7) << 2;
    uint2 pa; pa.x = pack_bf16x2(ra[i].x, ra[i].y); pa.y = pack_bf16x2(ra[i].z, ra[i].w);
    *(uint2*)(&As[r][c4]) = pa;
    uint2 pb; pb.x = pack_bf16x2(rb[i].x, rb[i].y); pb.y = pack_bf16x2(rb[i].z, rb[i].w);
    *(uint2*)(&Bs[r][c4]) = pb;
  }
}

// ---- branchless guarded staging (K remainder / partial N tile) -------------
// Clamp addresses into the valid row range (always mapped) and select zero
// with cndmask instead of predicating the loads: no exec-mask branches.
__device__ __forceinline__ void stage_slow(
    const float* __restrict__ A, const float* __restrict__ W,
    int N, int K, int rowBase, int colBase, int kBase, int tid,
    unsigned short (*As)[LDS_STRIDE], unsigned short (*Bs)[LDS_STRIDE]) {
#pragma unroll
  for (int i = 0; i < 16; ++i) {
    int idx = tid + i * 128;            // 0..2047
    int r = idx >> 5, c = idx & 31;
    int gc = kBase + c;
    int gcc = (gc < K) ? gc : (K - 1);                 // clamped, in-bounds
    float av = A[(long)(rowBase + r) * K + gcc];
    av = (gc < K) ? av : 0.0f;
    As[r][c] = f32_to_bf16(av);
    int gn = colBase + r;
    int gnn = (gn < N) ? gn : (N - 1);                 // clamped, in-bounds
    float wv = W[(long)gnn * K + gcc];
    wv = (gn < N && gc < K) ? wv : 0.0f;
    Bs[r][c] = f32_to_bf16(wv);
  }
}

// ---------------------------------------------------------------------------
// Generic GEMM: C[M,N] = act(A[M,K] @ W[N,K]^T + bias[N])
// f32 in memory, bf16 in LDS, f32 WMMA accumulators.
// Block = 128 threads (4 waves). Block tile 64x64, K step 32.
// Wave w computes rows [w*16, w*16+16) x all 64 cols (4 WMMAs / k-step).
// Pipelined double-buffer: issue global_load_b128 for tile k+1, run the
// WMMAs for tile k (hiding load latency), then convert+ds_store, one barrier.
// ---------------------------------------------------------------------------
__global__ __launch_bounds__(128) void gemm_bf16_kernel(
    const float* __restrict__ A, const float* __restrict__ W,
    const float* __restrict__ bias, float* __restrict__ C,
    float* __restrict__ Mask, int M, int N, int K, int flags) {
  __shared__ unsigned short As[2][BM][LDS_STRIDE];
  __shared__ unsigned short Bs[2][BN][LDS_STRIDE];

  const int tid  = threadIdx.x;
  const int wave = tid >> 5;
  const int lane = tid & 31;
  const int rowBase = blockIdx.y * BM;
  const int colBase = blockIdx.x * BN;
  const bool nfull = (colBase + BN) <= N;   // block-uniform

  floatx8 acc[4] = {};
  float4 ra[4], rb[4];

  const int ksteps = (K + BK - 1) / BK;
  // prologue: stage tile 0
  if (nfull && BK <= K) {
    load_phase(A, W, K, rowBase, colBase, 0, tid, ra, rb);
    store_phase(tid, ra, rb, As[0], Bs[0]);
  } else {
    stage_slow(A, W, N, K, rowBase, colBase, 0, tid, As[0], Bs[0]);
  }
  __syncthreads();

  for (int ks = 0; ks < ksteps; ++ks) {
    const int cur = ks & 1;
    const bool hasNext  = (ks + 1) < ksteps;
    const bool nextFast = hasNext && nfull && ((ks + 2) * BK) <= K;

    if (nextFast)   // issue next tile's global loads before compute
      load_phase(A, W, K, rowBase, colBase, (ks + 1) * BK, tid, ra, rb);

    const int hi = lane >> 4;
    bf16x16 afrag = load_frag(&As[cur][0][0], (wave << 4) + (lane & 15), hi);
#pragma unroll
    for (int n = 0; n < 4; ++n) {
      bf16x16 bfrag = load_frag(&Bs[cur][0][0], (n << 4) + (lane & 15), hi);
      acc[n] = __builtin_amdgcn_wmma_f32_16x16x32_bf16(
          false, afrag, false, bfrag, (short)0, acc[n], false, false);
    }

    if (hasNext) {
      if (nextFast)
        store_phase(tid, ra, rb, As[cur ^ 1], Bs[cur ^ 1]);
      else
        stage_slow(A, W, N, K, rowBase, colBase, (ks + 1) * BK, tid,
                   As[cur ^ 1], Bs[cur ^ 1]);
    }
    __syncthreads();
  }

  // Epilogue. C/D layout: VGPR r, lanes<16 -> M=r, lanes>=16 -> M=r+8; N=lane&15.
  const bool relu   = flags & FLAG_RELU;
  const bool storec = flags & FLAG_STOREC;
  const bool storem = flags & FLAG_MASK;
  const int mrow0 = rowBase + (wave << 4) + 8 * (lane >> 4);
  const int ncol0 = colBase + (lane & 15);
#pragma unroll
  for (int n = 0; n < 4; ++n) {
    int ncol = ncol0 + (n << 4);
    if (ncol >= N) continue;
    float bv = bias ? bias[ncol] : 0.0f;
#pragma unroll
    for (int r = 0; r < 8; ++r) {
      int mrow = mrow0 + r;
      float v = acc[n][r] + bv;
      if (storem) Mask[(long)mrow * N + ncol] = (v > 0.0f) ? 1.0f : 0.0f;
      if (relu) v = (v > 0.0f) ? v : 0.0f;
      if (storec) C[(long)mrow * N + ncol] = v;
    }
  }
}

// ---------------------------------------------------------------------------
// G[i,j] = sum_b m2[b,i] * m1[b,j]  (both operands transposed vs row-major)
// Split-K over B with f32 atomic reduction. Dim = 256. L2-resident traffic.
// ---------------------------------------------------------------------------
__global__ __launch_bounds__(128) void gemm_tt_atomic_kernel(
    const float* __restrict__ m2, const float* __restrict__ m1,
    float* __restrict__ G, int Dim, int kChunk) {
  __shared__ unsigned short As[BM][LDS_STRIDE];
  __shared__ unsigned short Bs[BN][LDS_STRIDE];

  const int tid  = threadIdx.x;
  const int wave = tid >> 5;
  const int lane = tid & 31;
  const int iBase = blockIdx.y * BM;
  const int jBase = blockIdx.x * BN;
  const int k0 = blockIdx.z * kChunk;

  floatx8 acc[4] = {};

  for (int ks = 0; ks < kChunk / BK; ++ks) {
    const int kBase = k0 + ks * BK;
#pragma unroll
    for (int i = 0; i < 16; ++i) {
      int idx = tid + i * 128;
      int il = idx & 63, kb = idx >> 6;     // coalesced along feature dim
      long gb = (long)(kBase + kb) * Dim;
      As[il][kb] = f32_to_bf16(m2[gb + iBase + il]);
      Bs[il][kb] = f32_to_bf16(m1[gb + jBase + il]);
    }
    __syncthreads();

    const int hi = lane >> 4;
    bf16x16 afrag = load_frag(&As[0][0], (wave << 4) + (lane & 15), hi);
#pragma unroll
    for (int n = 0; n < 4; ++n) {
      bf16x16 bfrag = load_frag(&Bs[0][0], (n << 4) + (lane & 15), hi);
      acc[n] = __builtin_amdgcn_wmma_f32_16x16x32_bf16(
          false, afrag, false, bfrag, (short)0, acc[n], false, false);
    }
    __syncthreads();
  }

  const int i0 = iBase + (wave << 4) + 8 * (lane >> 4);
  const int j0 = jBase + (lane & 15);
#pragma unroll
  for (int n = 0; n < 4; ++n)
#pragma unroll
    for (int r = 0; r < 8; ++r)
      atomicAdd(&G[(long)(i0 + r) * Dim + j0 + (n << 4)], acc[n][r]);
}

// ---------------------------------------------------------------------------
// Small helper kernels
// ---------------------------------------------------------------------------
__global__ void fill_kernel(float* __restrict__ p, int n, float v) {
  int i = blockIdx.x * blockDim.x + threadIdx.x;
  if (i < n) p[i] = v;
}

// T1[i,r] = sum_j (W2[i,j]*G[i,j]/B) * W1[j, 720 + r]     (256 x 360)
__global__ void t1_kernel(const float* __restrict__ W2, const float* __restrict__ G,
                          const float* __restrict__ W1, float* __restrict__ T1) {
  int idx = blockIdx.x * blockDim.x + threadIdx.x;
  if (idx >= Hh * Rr) return;
  int i = idx / Rr, r = idx % Rr;
  const float invB = 1.0f / (float)Bsz;
  float s = 0.0f;
  for (int j = 0; j < Hh; ++j)
    s += W2[i * Hh + j] * G[i * Hh + j] * invB * W1[(long)j * K3R + (K3R - Rr) + r];
  T1[idx] = s;
}

// ec[q,r] = sum_i W3[q,i] * T1[i,r]        (360 x 360)
__global__ void ec_kernel(const float* __restrict__ W3, const float* __restrict__ T1,
                          float* __restrict__ ec) {
  int idx = blockIdx.x * blockDim.x + threadIdx.x;
  if (idx >= Rr * Rr) return;
  int q = idx / Rr, r = idx % Rr;
  float s = 0.0f;
  for (int i = 0; i < Hh; ++i) s += W3[q * Hh + i] * T1[i * Rr + r];
  ec[idx] = s;
}

// p[j] += partial dot(ec_flat, Wp1[j, chunk]); float4 loads stream the 265 MB
// Wp1 at full width, with a software prefetch a few iterations ahead.
__global__ __launch_bounds__(256) void npi_dot_kernel(
    const float* __restrict__ ec, const float* __restrict__ Wp1,
    float* __restrict__ p, int chunk4) {
  __shared__ float red[256];
  const int j = blockIdx.x;
  const int n4 = RRf / 4;                       // 32400 float4 per row
  int start = blockIdx.y * chunk4;
  int end = start + chunk4; if (end > n4) end = n4;
  const float4* wrow = (const float4*)(Wp1 + (long)j * RRf);
  const float4* ec4  = (const float4*)ec;
  float s = 0.0f;
  for (int idx = start + threadIdx.x; idx < end; idx += 256) {
    __builtin_prefetch(wrow + idx + 512, 0, 1);
    float4 w = wrow[idx];
    float4 e = ec4[idx];
    s += e.x * w.x + e.y * w.y + e.z * w.z + e.w * w.w;
  }
  red[threadIdx.x] = s;
  __syncthreads();
  for (int off = 128; off > 0; off >>= 1) {
    if (threadIdx.x < off) red[threadIdx.x] += red[threadIdx.x + off];
    __syncthreads();
  }
  if (threadIdx.x == 0) atomicAdd(&p[j], red[0]);
}

// z_npi1[l] = bp2[l] + sum_j relu(p[j]) * Wp2[l,j]      (128)
__global__ void npi_proj2_kernel(const float* __restrict__ p, const float* __restrict__ Wp2,
                                 const float* __restrict__ bp2, float* __restrict__ znpi1) {
  int l = threadIdx.x;
  if (l >= LATl) return;
  float s = bp2[l];
  for (int j = 0; j < P1p; ++j) {
    float pj = p[j];
    s += (pj > 0.0f ? pj : 0.0f) * Wp2[l * P1p + j];
  }
  znpi1[l] = s;
}

// One wave32 per row: zsum = z_geo + z_npi1; zn = zsum / max(||zsum||, 1e-12)
// Writes zn and the broadcast z_npi output.
__global__ __launch_bounds__(256) void norm_kernel(
    const float* __restrict__ zgeo, const float* __restrict__ znpi1,
    float* __restrict__ out_zn, float* __restrict__ out_znpi) {
  int wave = threadIdx.x >> 5, lane = threadIdx.x & 31;
  long row = (long)blockIdx.x * 8 + wave;
  int c0 = lane * 4;
  float zs[4]; float ss = 0.0f;
#pragma unroll
  for (int k = 0; k < 4; ++k) {
    float z = zgeo[row * LATl + c0 + k] + znpi1[c0 + k];
    zs[k] = z; ss += z * z;
  }
#pragma unroll
  for (int off = 16; off > 0; off >>= 1) ss += __shfl_xor(ss, off, 32);
  float nrm = sqrtf(ss);
  float scale = 1.0f / fmaxf(nrm, 1e-12f);
#pragma unroll
  for (int k = 0; k < 4; ++k) {
    out_zn[row * LATl + c0 + k]   = zs[k] * scale;
    out_znpi[row * LATl + c0 + k] = znpi1[c0 + k];
  }
}

// ---------------------------------------------------------------------------
// Launch
// ---------------------------------------------------------------------------
extern "C" void kernel_launch(void* const* d_in, const int* in_sizes, int n_in,
                              void* d_out, int out_size, void* d_ws, size_t ws_size,
                              hipStream_t stream) {
  (void)in_sizes; (void)n_in; (void)out_size; (void)ws_size;
  const float* x      = (const float*)d_in[0];
  // d_in[1] = modes (unused by reference)
  const float* mcoef  = (const float*)d_in[2];
  const float* W1 = (const float*)d_in[3];  const float* b1 = (const float*)d_in[4];
  const float* W2 = (const float*)d_in[5];  const float* b2 = (const float*)d_in[6];
  const float* W3 = (const float*)d_in[7];  // b3 unused
  const float* Wp1 = (const float*)d_in[9];  const float* bp1 = (const float*)d_in[10];
  const float* Wp2 = (const float*)d_in[11]; const float* bp2 = (const float*)d_in[12];
  const float* Wg1 = (const float*)d_in[13]; const float* bg1 = (const float*)d_in[14];
  const float* Wg2 = (const float*)d_in[15]; const float* bg2 = (const float*)d_in[16];
  const float* Wr1 = (const float*)d_in[17]; const float* br1 = (const float*)d_in[18];
  const float* Wr2 = (const float*)d_in[19]; const float* br2 = (const float*)d_in[20];

  float* out = (float*)d_out;
  // output layout: z_npi | z_geo | zn | ec | recon
  float* o_znpi  = out;
  float* o_zgeo  = out + (long)Bsz * LATl;
  float* o_zn    = out + 2L * Bsz * LATl;
  float* o_ec    = out + 3L * Bsz * LATl;
  float* o_recon = o_ec + RRf;

  float* ws = (float*)d_ws;
  float* w_h1r   = ws;                       // B x 256
  float* w_m1    = w_h1r + (long)Bsz * Hh;   // B x 256
  float* w_m2    = w_m1  + (long)Bsz * Hh;   // B x 256
  float* w_zg1   = w_m2  + (long)Bsz * Hh;   // B x 256
  float* w_r1    = w_zg1 + (long)Bsz * Hh;   // B x 256
  float* w_G     = w_r1  + (long)Bsz * Hh;   // 256 x 256
  float* w_T1    = w_G   + Hh * Hh;          // 256 x 360
  float* w_p     = w_T1  + Hh * Rr;          // 512
  float* w_znpi1 = w_p   + P1p;              // 128

  const dim3 blk(128);
  const int MT = Bsz / BM;                   // 256 row tiles

  // 1) h1r = relu(x @ W1^T + b1), m1 = (h1 > 0)
  gemm_bf16_kernel<<<dim3(Hh / BN, MT), blk, 0, stream>>>(
      x, W1, b1, w_h1r, w_m1, Bsz, Hh, K3R, FLAG_RELU | FLAG_STOREC | FLAG_MASK);

  // 2) m2 = (h1r @ W2^T + b2 > 0)   (mask only)
  gemm_bf16_kernel<<<dim3(Hh / BN, MT), blk, 0, stream>>>(
      w_h1r, W2, b2, nullptr, w_m2, Bsz, Hh, Hh, FLAG_MASK);

  // 3) G = m2^T @ m1  (split-K atomic)
  fill_kernel<<<(Hh * Hh + 255) / 256, 256, 0, stream>>>(w_G, Hh * Hh, 0.0f);
  gemm_tt_atomic_kernel<<<dim3(Hh / BN, Hh / BM, 16), blk, 0, stream>>>(
      w_m2, w_m1, w_G, Hh, Bsz / 16);

  // 4) T1 = (W2*G/B) @ W1[:, -R:] ; ec = W3 @ T1
  t1_kernel<<<(Hh * Rr + 255) / 256, 256, 0, stream>>>(W2, w_G, W1, w_T1);
  ec_kernel<<<(RRf + 255) / 256, 256, 0, stream>>>(W3, w_T1, o_ec);

  // 5) npi projection: p = bp1 + Wp1 @ ec_flat; z_npi1 = relu(p) @ Wp2^T + bp2
  hipMemcpyAsync(w_p, bp1, P1p * sizeof(float), hipMemcpyDeviceToDevice, stream);
  {
    const int NSPLIT = 16;
    const int n4 = RRf / 4;
    const int chunk4 = (n4 + NSPLIT - 1) / NSPLIT;
    npi_dot_kernel<<<dim3(P1p, NSPLIT), 256, 0, stream>>>(o_ec, Wp1, w_p, chunk4);
  }
  npi_proj2_kernel<<<1, 128, 0, stream>>>(w_p, Wp2, bp2, w_znpi1);

  // 6) z_geo = relu(mcoef @ Wg1^T + bg1) @ Wg2^T + bg2
  gemm_bf16_kernel<<<dim3(Hh / BN, MT), blk, 0, stream>>>(
      mcoef, Wg1, bg1, w_zg1, nullptr, Bsz, Hh, NMm, FLAG_RELU | FLAG_STOREC);
  gemm_bf16_kernel<<<dim3(LATl / BN, MT), blk, 0, stream>>>(
      w_zg1, Wg2, bg2, o_zgeo, nullptr, Bsz, LATl, Hh, FLAG_STOREC);

  // 7) zn = normalize(z_geo + z_npi1); also broadcast z_npi
  norm_kernel<<<Bsz / 8, 256, 0, stream>>>(o_zgeo, w_znpi1, o_zn, o_znpi);

  // 8) recon = relu(zn @ Wr1^T + br1) @ Wr2^T + br2
  gemm_bf16_kernel<<<dim3(Hh / BN, MT), blk, 0, stream>>>(
      o_zn, Wr1, br1, w_r1, nullptr, Bsz, Hh, LATl, FLAG_RELU | FLAG_STOREC);
  gemm_bf16_kernel<<<dim3((Rr + BN - 1) / BN, MT), blk, 0, stream>>>(
      w_r1, Wr2, br2, o_recon, nullptr, Bsz, Rr, Hh, FLAG_STOREC);
}
